// SelfConditionedMultiheadAttention_8065948582239
// MI455X (gfx1250) — matrix operand order repaired and checked by reading, not compiled, not run on hardware
//
#include <hip/hip_runtime.h>
#include <hip/hip_bf16.h>

// Problem constants (match reference)
#define L_SEQ 2048
#define B_SZ  2
#define E_DIM 1024
#define H_N   16
#define D_HD  64
#define R_DIM 256

typedef __attribute__((ext_vector_type(16))) __bf16 v16bf;
typedef __attribute__((ext_vector_type(8)))  __bf16 v8bf;
typedef __attribute__((ext_vector_type(8)))  float  v8f;

// ---------------------------------------------------------------------------
// Generic batched GEMM: C[z] = alpha * (A[z] x B[z]) + bias + beta * Res[z]
//   A: f32 row-major [M x K]; B: f32 (k,n) at k*ldb+n, or n*ldb+k if btrans.
//   All launches in this problem have M,N % 64 == 0 and K % 32 == 0, and all
//   leading dims % 8 == 0, so there are NO bounds guards (keeps EXEC all-ones
//   and lets loads issue wide and unpredicated).
// Tiling: 64x64 block, BLOCK_K=32, 256 threads = 8 waves (4x2 wave grid),
// each wave computes two 16x16 tiles with v_wmma_f32_16x16x32_bf16.
// LDS: As[m][k] and Bs2[n][k] with 96-byte row pitch so every WMMA fragment
// is two contiguous 16-byte chunks -> ds_load_b128.
// ---------------------------------------------------------------------------
struct GemmP {
  const float* A;  long lda;  long aOffB; long aOffH;
  const float* Bm; long ldb;  long bOffB; long bOffH; int btrans;
  float* C;        long ldc;  long cOffB; long cOffH;
  const float* bias;                                   // per-n bias or null
  const float* Res; long ldres; long rOffB; long rOffH; // residual or null
  float alpha, beta;
  int M, N, K, Hdiv;
};

__global__ __launch_bounds__(256) void gemm_wmma_bf16(GemmP p) {
  const int z  = blockIdx.z;
  const int zb = z / p.Hdiv;
  const int zh = z % p.Hdiv;
  const float* A  = p.A  + (long)zb * p.aOffB + (long)zh * p.aOffH;
  const float* Bm = p.Bm + (long)zb * p.bOffB + (long)zh * p.bOffH;
  float*       C  = p.C  + (long)zb * p.cOffB + (long)zh * p.cOffH;
  const float* Res = p.Res ? (p.Res + (long)zb * p.rOffB + (long)zh * p.rOffH)
                           : nullptr;

  // Row pitch 48 bf16 = 96 bytes: multiple of 16B (aligned b128 LDS ops).
  __shared__ __align__(16) __bf16 As [64][48];  // [m][k], k in 0..31
  __shared__ __align__(16) __bf16 Bs2[64][48];  // [n][k], k in 0..31

  const int m0   = blockIdx.y << 6;
  const int n0   = blockIdx.x << 6;
  const int tid  = threadIdx.x;
  const int lane = tid & 31;
  const int wave = tid >> 5;
  const int wm   = wave & 3;   // tile row (0..3)
  const int wn   = wave >> 2;  // tile-pair col (0..1)

  v8f acc0 = {};
  v8f acc1 = {};

  // Per-thread staging coordinates (one 8-float run each => 2x b128 loads).
  const int ar = tid >> 2, ac8 = (tid & 3) << 3;        // A: 64 rows x 32 k
  const int tn = tid >> 2, tk8 = (tid & 3) << 3;        // B btrans: 64 n x 32 k
  const int nk = tid >> 3, nn8 = (tid & 7) << 3;        // B direct: 32 k x 64 n

  for (int k0 = 0; k0 < p.K; k0 += 32) {
    // ---- stage A (64x32): 8 consecutive f32 per thread -> one b128 LDS store
    {
      const float4* g = (const float4*)(A + (long)(m0 + ar) * p.lda + k0 + ac8);
      float4 a0 = g[0], a1 = g[1];
      v8bf pk = {(__bf16)a0.x, (__bf16)a0.y, (__bf16)a0.z, (__bf16)a0.w,
                 (__bf16)a1.x, (__bf16)a1.y, (__bf16)a1.z, (__bf16)a1.w};
      *(v8bf*)&As[ar][ac8] = pk;
    }
    // ---- stage B into Bs2[n][k] ----
    if (p.btrans) {
      // memory (n,k) contiguous in k: coalesced loads, contiguous LDS store
      const float4* g = (const float4*)(Bm + (long)(n0 + tn) * p.ldb + k0 + tk8);
      float4 b0 = g[0], b1 = g[1];
      v8bf pk = {(__bf16)b0.x, (__bf16)b0.y, (__bf16)b0.z, (__bf16)b0.w,
                 (__bf16)b1.x, (__bf16)b1.y, (__bf16)b1.z, (__bf16)b1.w};
      *(v8bf*)&Bs2[tn][tk8] = pk;
    } else {
      // memory (k,n) contiguous in n: coalesced loads, transposed b16 stores
      const float4* g = (const float4*)(Bm + (long)(k0 + nk) * p.ldb + n0 + nn8);
      float4 b0 = g[0], b1 = g[1];
      Bs2[nn8 + 0][nk] = (__bf16)b0.x;
      Bs2[nn8 + 1][nk] = (__bf16)b0.y;
      Bs2[nn8 + 2][nk] = (__bf16)b0.z;
      Bs2[nn8 + 3][nk] = (__bf16)b0.w;
      Bs2[nn8 + 4][nk] = (__bf16)b1.x;
      Bs2[nn8 + 5][nk] = (__bf16)b1.y;
      Bs2[nn8 + 6][nk] = (__bf16)b1.z;
      Bs2[nn8 + 7][nk] = (__bf16)b1.w;
    }
    // prefetch next A K-tile (global_prefetch_b8)
    if (k0 + 32 < p.K) {
      __builtin_prefetch(A + (long)(m0 + ar) * p.lda + (k0 + 32) + ac8, 0, 1);
    }
    __syncthreads();

    // ---- WMMA fragments (ISA 7.12.2 layouts), contiguous b128 LDS loads ----
    // A 16x32: lane -> M = lane%16, kg = lane/16; lane's K-runs are
    //   [8kg..8kg+7] then [16+8kg..16+8kg+7]  -> 2 x 16B chunks.
    const int kg   = lane >> 4;
    const int mrow = (wm << 4) + (lane & 15);
    const v8bf* aRow = (const v8bf*)&As[mrow][0];
    v8bf a_lo = aRow[kg];          // bf16 offset 8*kg
    v8bf a_hi = aRow[2 + kg];      // bf16 offset 16 + 8*kg
    v16bf af = __builtin_shufflevector(a_lo, a_hi,
        0, 1, 2, 3, 4, 5, 6, 7, 8, 9, 10, 11, 12, 13, 14, 15);
    // B 32x16: lane -> N = lane%16; K-runs [8kg..8kg+7] then [16+8kg..16+8kg+7].
    const int ncol = (wn << 5) + (lane & 15);
    const v8bf* bRow0 = (const v8bf*)&Bs2[ncol][0];
    const v8bf* bRow1 = (const v8bf*)&Bs2[ncol + 16][0];
    v16bf bfr0 = __builtin_shufflevector(bRow0[kg], bRow0[2 + kg],
        0, 1, 2, 3, 4, 5, 6, 7, 8, 9, 10, 11, 12, 13, 14, 15);
    v16bf bfr1 = __builtin_shufflevector(bRow1[kg], bRow1[2 + kg],
        0, 1, 2, 3, 4, 5, 6, 7, 8, 9, 10, 11, 12, 13, 14, 15);

    acc0 = __builtin_amdgcn_wmma_f32_16x16x32_bf16(
        false, af, false, bfr0, (short)0, acc0, false, false);
    acc1 = __builtin_amdgcn_wmma_f32_16x16x32_bf16(
        false, af, false, bfr1, (short)0, acc1, false, false);
    __syncthreads();
  }

  // ---- epilogue: C/D 16x16 f32: N = lane%16, M = i + 8*(lane/16) ----
  const int crow = (lane >> 4) << 3;
  const int ccol = lane & 15;
#pragma unroll
  for (int i = 0; i < 8; ++i) {
    const int gm  = m0 + (wm << 4) + crow + i;
    const int gn0 = n0 + (wn << 5) + ccol;
    const int gn1 = gn0 + 16;
    float v0 = p.alpha * acc0[i];
    float v1 = p.alpha * acc1[i];
    if (p.bias) { v0 += p.bias[gn0]; v1 += p.bias[gn1]; }
    if (Res) {
      v0 += p.beta * Res[(long)gm * p.ldres + gn0];
      v1 += p.beta * Res[(long)gm * p.ldres + gn1];
    }
    C[(long)gm * p.ldc + gn0] = v0;
    C[(long)gm * p.ldc + gn1] = v1;
  }
}

// ---------------------------------------------------------------------------
// In-place row softmax: one 256-thread block per row of length n (=2048).
// ---------------------------------------------------------------------------
__global__ __launch_bounds__(256) void softmax_rows(float* attn, int n) {
  const long row = blockIdx.x;
  float* p = attn + row * (long)n;
  __shared__ float red[256];
  const int tid = threadIdx.x;

  float mx = -3.4e38f;
  for (int i = tid; i < n; i += 256) mx = fmaxf(mx, p[i]);
  red[tid] = mx;
  __syncthreads();
  for (int s = 128; s > 0; s >>= 1) {
    if (tid < s) red[tid] = fmaxf(red[tid], red[tid + s]);
    __syncthreads();
  }
  const float m = red[0];
  __syncthreads();

  float sum = 0.0f;
  for (int i = tid; i < n; i += 256) {
    float e = __expf(p[i] - m);
    p[i] = e;
    sum += e;
  }
  red[tid] = sum;
  __syncthreads();
  for (int s = 128; s > 0; s >>= 1) {
    if (tid < s) red[tid] += red[tid + s];
    __syncthreads();
  }
  const float inv = 1.0f / red[0];
  for (int i = tid; i < n; i += 256) p[i] *= inv;
}

// ---------------------------------------------------------------------------
// Host-side orchestration
// ---------------------------------------------------------------------------
extern "C" void kernel_launch(void* const* d_in, const int* in_sizes, int n_in,
                              void* d_out, int out_size, void* d_ws, size_t ws_size,
                              hipStream_t stream) {
  const float* x    = (const float*)d_in[0];
  const float* mask = (const float*)d_in[1];
  const float* Wq   = (const float*)d_in[2];
  const float* bq   = (const float*)d_in[3];
  const float* Wk   = (const float*)d_in[4];
  const float* bk   = (const float*)d_in[5];
  const float* Wv   = (const float*)d_in[6];
  const float* bv   = (const float*)d_in[7];
  const float* Wo   = (const float*)d_in[8];
  const float* bo   = (const float*)d_in[9];
  const float* U    = (const float*)d_in[10];

  float* out  = (float*)d_out;                              // [L,B,E]
  float* attn = out + (size_t)L_SEQ * B_SZ * E_DIM;         // [B,H,L,L]

  // Workspace layout (f32). NTOK = L*B = 4096 token rows.
  const size_t NTOK = (size_t)L_SEQ * B_SZ;
  float* ws = (float*)d_ws;
  float* Q   = ws;                       // [NTOK, E]
  float* Kb  = Q  + NTOK * E_DIM;        // [NTOK, E]
  float* V   = Kb + NTOK * E_DIM;        // [NTOK, E]
  float* Pq  = V  + NTOK * E_DIM;        // [NTOK, R]
  float* Pk  = Pq + NTOK * R_DIM;        // [NTOK, R]
  float* Qc  = Pk + NTOK * R_DIM;        // [NTOK, E]
  float* Kc  = Qc + NTOK * E_DIM;        // [NTOK, E]
  float* ctx = Q;                        // reuse Q region (free after Qc)

  const long LLL = (long)L_SEQ * L_SEQ;  // per-head attn slice

  auto run = [&](GemmP p, int gx, int gy, int gz) {
    gemm_wmma_bf16<<<dim3(gx, gy, gz), 256, 0, stream>>>(p);
  };

  GemmP p{};

  // ---- Q = x @ Wq.T + bq ----
  p = {}; p.A = x; p.lda = E_DIM; p.Bm = Wq; p.ldb = E_DIM; p.btrans = 1;
  p.C = Q; p.ldc = E_DIM; p.bias = bq; p.alpha = 1.0f; p.beta = 0.0f;
  p.M = (int)NTOK; p.N = E_DIM; p.K = E_DIM; p.Hdiv = 1;
  run(p, E_DIM / 64, (int)NTOK / 64, 1);

  // ---- K = x @ Wk.T + bk ----
  p.Bm = Wk; p.bias = bk; p.C = Kb;
  run(p, E_DIM / 64, (int)NTOK / 64, 1);

  // ---- V = x @ Wv.T + bv ----
  p.Bm = Wv; p.bias = bv; p.C = V;
  run(p, E_DIM / 64, (int)NTOK / 64, 1);

  // ---- Pq = Q @ U ----
  p = {}; p.A = Q; p.lda = E_DIM; p.Bm = U; p.ldb = R_DIM; p.btrans = 0;
  p.C = Pq; p.ldc = R_DIM; p.alpha = 1.0f;
  p.M = (int)NTOK; p.N = R_DIM; p.K = E_DIM; p.Hdiv = 1;
  run(p, R_DIM / 64, (int)NTOK / 64, 1);

  // ---- Pk = K @ U ----
  p.A = Kb; p.C = Pk;
  run(p, R_DIM / 64, (int)NTOK / 64, 1);

  // ---- Qc = 0.4 * (Pq @ U.T) + 0.6 * Q ----
  p = {}; p.A = Pq; p.lda = R_DIM; p.Bm = U; p.ldb = R_DIM; p.btrans = 1;
  p.C = Qc; p.ldc = E_DIM; p.alpha = 0.4f;
  p.Res = Q; p.ldres = E_DIM; p.beta = 0.6f;
  p.M = (int)NTOK; p.N = E_DIM; p.K = R_DIM; p.Hdiv = 1;
  run(p, E_DIM / 64, (int)NTOK / 64, 1);

  // ---- Kc = 0.4 * (Pk @ U.T) + 0.6 * K ----
  p.A = Pk; p.Res = Kb; p.C = Kc;
  run(p, E_DIM / 64, (int)NTOK / 64, 1);

  // ---- scores[b,h] = (1/sqrt(D)) * Qc_bh @ Kc_bh.T + mask  (into d_out) ----
  p = {};
  p.A = Qc; p.lda = (long)B_SZ * E_DIM; p.aOffB = E_DIM; p.aOffH = D_HD;
  p.Bm = Kc; p.ldb = (long)B_SZ * E_DIM; p.bOffB = E_DIM; p.bOffH = D_HD; p.btrans = 1;
  p.C = attn; p.ldc = L_SEQ; p.cOffB = (long)H_N * LLL; p.cOffH = LLL;
  p.Res = mask; p.ldres = L_SEQ; p.rOffB = 0; p.rOffH = 0;
  p.alpha = 0.125f;  // 1/sqrt(64)
  p.beta = 1.0f;
  p.M = L_SEQ; p.N = L_SEQ; p.K = D_HD; p.Hdiv = H_N;
  run(p, L_SEQ / 64, L_SEQ / 64, B_SZ * H_N);

  // ---- softmax over last dim, in place in d_out attn region ----
  softmax_rows<<<dim3(B_SZ * H_N * L_SEQ), 256, 0, stream>>>(attn, L_SEQ);

  // ---- ctx[b,h] = attn_bh @ V_bh ----
  p = {};
  p.A = attn; p.lda = L_SEQ; p.aOffB = (long)H_N * LLL; p.aOffH = LLL;
  p.Bm = V; p.ldb = (long)B_SZ * E_DIM; p.bOffB = E_DIM; p.bOffH = D_HD; p.btrans = 0;
  p.C = ctx; p.ldc = (long)B_SZ * E_DIM; p.cOffB = E_DIM; p.cOffH = D_HD;
  p.alpha = 1.0f;
  p.M = L_SEQ; p.N = D_HD; p.K = L_SEQ; p.Hdiv = H_N;
  run(p, 1, L_SEQ / 64, B_SZ * H_N);

  // ---- out = ctx @ Wo.T + bo  (ctx is token-major [NTOK, E]) ----
  p = {};
  p.A = ctx; p.lda = E_DIM; p.Bm = Wo; p.ldb = E_DIM; p.btrans = 1;
  p.C = out; p.ldc = E_DIM; p.bias = bo; p.alpha = 1.0f;
  p.M = (int)NTOK; p.N = E_DIM; p.K = E_DIM; p.Hdiv = 1;
  run(p, E_DIM / 64, (int)NTOK / 64, 1);
}
